// TTSModel_87548613361984
// MI455X (gfx1250) — matrix operand order repaired and checked by reading, not compiled, not run on hardware
//
#include <hip/hip_runtime.h>

// ---------------------------------------------------------------------------
// Tacotron-2 style decoder for MI455X (gfx1250, wave32, WMMA).
//   - f16 copies of enc_outputs + attn_enc (67MB total) stay resident in 192MB L2.
//   - All GEMMs via v_wmma_f32_16x16x32_f16 with K-major pre-transposed weights.
//   - Per-step: k_attn (64 WGs x 1024 thr, one WG per batch row)
//               -> k_lstm (16 WGs, WMMA gates + local cell update).
//   - Prenet precomputed (batched WMMA), heads batched at end (WMMA).
//   - tanh via hardware v_tanh_f32 (TRANS) in the dominant attention loop.
// ---------------------------------------------------------------------------

#define T_OUTS 1000

typedef __attribute__((ext_vector_type(16))) _Float16 v16h;
typedef __attribute__((ext_vector_type(8)))  _Float16 v8h;
typedef __attribute__((ext_vector_type(8)))  float    v8f;

#define DEV static __device__ __forceinline__

DEV float fast_tanh(float x) {
#if __has_builtin(__builtin_amdgcn_tanhf)
  return __builtin_amdgcn_tanhf(x);          // gfx1250 v_tanh_f32
#else
  float r;
  asm("v_tanh_f32 %0, %1" : "=v"(r) : "v"(x));
  return r;
#endif
}
DEV float sigm(float x) { return 1.0f / (1.0f + __expf(-x)); }

DEV v8f wmma16(v16h a, v16h b, v8f c) {
  return __builtin_amdgcn_wmma_f32_16x16x32_f16(false, a, false, b, (short)0, c,
                                                false, false);
}

// A fragment (16x32 f16) from row-major [.. , lda] buffer, base pre-offset to (m0,k0).
// ISA layout: lanes<16 hold K 0..7 / 16..23, lanes>=16 hold K 8..15 / 24..31.
DEV v16h load_a(const _Float16* Abase, int lda, int lane) {
  const int m = lane & 15, hi = lane >> 4;
  const _Float16* p = Abase + (size_t)m * lda + hi * 8;
  union { v16h v; v8h h[2]; } u;
  u.h[0] = *(const v8h*)(p);
  u.h[1] = *(const v8h*)(p + 16);
  return u.v;
}
// B fragment (32x16 f16) from K-major Bt[ldK rows][ldb cols], base at (k0, n0):
// lane L supplies column n0+(L&15); K 0..7 / 16..23 (lanes<16), K 8..15 / 24..31 (lanes>=16).
DEV v16h load_b(const _Float16* Bt_k0_n0, int ldb, int lane) {
  const int n = lane & 15, hi = lane >> 4;
  v16h v;
#pragma unroll
  for (int s = 0; s < 8; ++s) {
    v[s]     = Bt_k0_n0[(size_t)(hi * 8 + s) * ldb + n];
    v[s + 8] = Bt_k0_n0[(size_t)(16 + hi * 8 + s) * ldb + n];
  }
  return v;
}

// ------------------------- weight preparation ------------------------------

__global__ void k_cvt(const float* __restrict__ S, _Float16* __restrict__ D, int n) {
  int i = blockIdx.x * 256 + threadIdx.x;
  if (i < n) D[i] = (_Float16)S[i];
}
// D[k][256] = W[n*256+k]  (256x256 transpose to K-major)
__global__ void k_t256(const float* __restrict__ W, _Float16* __restrict__ D) {
  int i = blockIdx.x * 256 + threadIdx.x;
  if (i < 256 * 256) { int k = i >> 8, n = i & 255; D[(size_t)k * 256 + n] = (_Float16)W[(size_t)n * 256 + k]; }
}
// Wcombt[768][1024]: k<512 -> W_ih[n][k], else W_hh[n][k-512]; bias = b_ih+b_hh
__global__ void k_wcomb(const float* __restrict__ Wih, const float* __restrict__ Whh,
                        const float* __restrict__ bih, const float* __restrict__ bhh,
                        _Float16* __restrict__ D, float* __restrict__ bias) {
  int i = blockIdx.x * 256 + threadIdx.x;
  if (i < 768 * 1024) {
    int k = i >> 10, n = i & 1023;
    float v = (k < 512) ? Wih[(size_t)n * 512 + k] : Whh[(size_t)n * 256 + (k - 512)];
    D[(size_t)k * 1024 + n] = (_Float16)v;
  }
  if (i < 1024) bias[i] = bih[i] + bhh[i];
}
// Wheadt[512][96]: n<80 -> mel_W[n][k], n==80 -> stop_W[k], else 0
__global__ void k_whead(const float* __restrict__ melW, const float* __restrict__ stopW,
                        const float* __restrict__ melb, const float* __restrict__ stopb,
                        _Float16* __restrict__ D, float* __restrict__ bias) {
  int i = blockIdx.x * 256 + threadIdx.x;
  if (i < 512 * 96) {
    int k = i / 96, n = i % 96;
    float v = 0.f;
    if (n < 80) v = melW[(size_t)n * 512 + k];
    else if (n == 80) v = stopW[k];
    D[(size_t)k * 96 + n] = (_Float16)v;
  }
  if (i < 96) bias[i] = (i < 80) ? melb[i] : (i == 80 ? stopb[0] : 0.f);
}
// W1t[96][256]: k<80 -> pre_W1[n][k], else 0 (zero-padded K for the 80-dim prenet input)
__global__ void k_w1t(const float* __restrict__ W1, _Float16* __restrict__ D) {
  int i = blockIdx.x * 256 + threadIdx.x;
  if (i < 96 * 256) {
    int k = i >> 8, n = i & 255;
    D[(size_t)k * 256 + n] = (_Float16)((k < 80) ? W1[(size_t)n * 80 + k] : 0.f);
  }
}
__global__ void k_init(_Float16* __restrict__ h0, float* __restrict__ c0) {
  int i = blockIdx.x * 256 + threadIdx.x;
  if (i < 64 * 256) { h0[i] = (_Float16)0.f; c0[i] = 0.f; }
}

// ----------------- encoder projection (batched WMMA GEMM) ------------------
// attn_enc16 = f16(enc @ W_enc^T + b_enc); also emits enc16 (f16 copy of enc).
__global__ __launch_bounds__(256) void k_encproj(const float* __restrict__ enc,
                                                 const float* __restrict__ b_enc,
                                                 const _Float16* __restrict__ Wenct,
                                                 _Float16* __restrict__ enc16,
                                                 _Float16* __restrict__ attnenc16) {
  __shared__ __align__(32) _Float16 A[64][256];
  const size_t R0 = (size_t)blockIdx.x * 64;   // 1024 blocks x 64 rows of (B*T_enc)
  for (int i = threadIdx.x; i < 64 * 256; i += 256) {
    int m = i >> 8, k = i & 255;
    _Float16 hv = (_Float16)enc[(R0 + m) * 256 + k];
    A[m][k] = hv;
    enc16[(R0 + m) * 256 + k] = hv;
  }
  __syncthreads();
  const int wave = threadIdx.x >> 5, lane = threadIdx.x & 31;
  const int nt0 = wave * 2;                          // wave handles n-tiles {2w,2w+1}
  for (int mi = 0; mi < 4; ++mi) {
    v8f acc0 = {}, acc1 = {};
    for (int kk = 0; kk < 8; ++kk) {
      v16h a  = load_a(&A[mi * 16][0] + kk * 32, 256, lane);
      v16h b0 = load_b(Wenct + (size_t)(kk * 32) * 256 + nt0 * 16, 256, lane);
      v16h b1 = load_b(Wenct + (size_t)(kk * 32) * 256 + (nt0 + 1) * 16, 256, lane);
      acc0 = wmma16(a, b0, acc0);
      acc1 = wmma16(a, b1, acc1);
    }
    const int nlo = lane & 15, hi = lane >> 4;
    for (int q = 0; q < 2; ++q) {
      const v8f acc = q ? acc1 : acc0;
      const int n = (nt0 + q) * 16 + nlo;
      const float bb = b_enc[n];
#pragma unroll
      for (int r = 0; r < 8; ++r) {
        int m = mi * 16 + r + 8 * hi;
        attnenc16[(R0 + m) * 256 + n] = (_Float16)(acc[r] + bb);
      }
    }
  }
}

// ------------------- prenet for all steps (batched WMMA) -------------------
__global__ __launch_bounds__(256) void k_prenet(const float* __restrict__ tmel,
                                                const float* __restrict__ b1,
                                                const float* __restrict__ b2,
                                                const _Float16* __restrict__ W1t,
                                                const _Float16* __restrict__ W2t,
                                                _Float16* __restrict__ pre_all) {
  __shared__ __align__(32) _Float16 A0[64][96];
  __shared__ __align__(32) _Float16 A1[64][256];
  const int t = blockIdx.x;
  for (int i = threadIdx.x; i < 64 * 96; i += 256) {
    int m = i / 96, k = i % 96;
    float v = 0.f;
    if (t > 0 && k < 80) v = tmel[((size_t)m * T_OUTS + (t - 1)) * 80 + k];
    A0[m][k] = (_Float16)v;
  }
  __syncthreads();
  const int wave = threadIdx.x >> 5, lane = threadIdx.x & 31;
  const int nt0 = wave * 2, nlo = lane & 15, hi = lane >> 4;
  // GEMM1: 64x256 = relu(A0(64x96) @ W1t + b1)
  for (int mi = 0; mi < 4; ++mi) {
    v8f acc0 = {}, acc1 = {};
    for (int kk = 0; kk < 3; ++kk) {
      v16h a  = load_a(&A0[mi * 16][0] + kk * 32, 96, lane);
      v16h b0 = load_b(W1t + (size_t)(kk * 32) * 256 + nt0 * 16, 256, lane);
      v16h bq = load_b(W1t + (size_t)(kk * 32) * 256 + (nt0 + 1) * 16, 256, lane);
      acc0 = wmma16(a, b0, acc0);
      acc1 = wmma16(a, bq, acc1);
    }
    for (int q = 0; q < 2; ++q) {
      const v8f acc = q ? acc1 : acc0;
      const int n = (nt0 + q) * 16 + nlo;
      const float bb = b1[n];
#pragma unroll
      for (int r = 0; r < 8; ++r) {
        int m = mi * 16 + r + 8 * hi;
        A1[m][n] = (_Float16)fmaxf(acc[r] + bb, 0.f);
      }
    }
  }
  __syncthreads();
  // GEMM2: pre = relu(A1 @ W2t + b2)
  for (int mi = 0; mi < 4; ++mi) {
    v8f acc0 = {}, acc1 = {};
    for (int kk = 0; kk < 8; ++kk) {
      v16h a  = load_a(&A1[mi * 16][0] + kk * 32, 256, lane);
      v16h b0 = load_b(W2t + (size_t)(kk * 32) * 256 + nt0 * 16, 256, lane);
      v16h bq = load_b(W2t + (size_t)(kk * 32) * 256 + (nt0 + 1) * 16, 256, lane);
      acc0 = wmma16(a, b0, acc0);
      acc1 = wmma16(a, bq, acc1);
    }
    for (int q = 0; q < 2; ++q) {
      const v8f acc = q ? acc1 : acc0;
      const int n = (nt0 + q) * 16 + nlo;
      const float bb = b2[n];
#pragma unroll
      for (int r = 0; r < 8; ++r) {
        int m = mi * 16 + r + 8 * hi;
        pre_all[((size_t)t * 64 + m) * 256 + n] = (_Float16)fmaxf(acc[r] + bb, 0.f);
      }
    }
  }
}

// ------------------- per-step attention (one WG per batch row) -------------
// 1024 threads (32 wave32s): one encoder position per thread for the energy
// pass; j-split partials for attn_dec; quarter-split context accumulation.
__global__ __launch_bounds__(1024) void k_attn(const _Float16* __restrict__ attnenc16,
                                               const _Float16* __restrict__ enc16,
                                               const _Float16* __restrict__ Wdec16,
                                               const _Float16* __restrict__ h_in,
                                               const float* __restrict__ b_dec,
                                               const float* __restrict__ v_w,
                                               const float* __restrict__ v_b,
                                               const float* __restrict__ enc_mask,
                                               _Float16* __restrict__ ctx_t) {
  __shared__ float h_s[256], d_s[256], v_s[256];
  __shared__ float e_s[1024];
  __shared__ float red[1024];
  const int b = blockIdx.x, tid = threadIdx.x;
  if (tid < 256) {
    h_s[tid] = (float)h_in[b * 256 + tid];
    v_s[tid] = v_w[tid];
  }
  __syncthreads();
  // attn_dec[a] = sum_j Wdec[a][j]*h[j] + b_dec[a]; 4 threads per output a,
  // each covering a 64-wide j-range.
  {
    const int a = tid & 255, q = tid >> 8;
    float acc = 0.f;
    const _Float16* wr = Wdec16 + (size_t)a * 256 + q * 64;
#pragma unroll 2
    for (int j = 0; j < 64; j += 8) {
      v8h w = *(const v8h*)(wr + j);
#pragma unroll
      for (int u = 0; u < 8; ++u) acc += (float)w[u] * h_s[q * 64 + j + u];
    }
    red[tid] = acc;
  }
  __syncthreads();
  if (tid < 256)
    d_s[tid] = red[tid] + red[tid + 256] + red[tid + 512] + red[tid + 768] + b_dec[tid];
  __syncthreads();
  // energy: one encoder position per thread (512B f16 row, L2-resident)
  {
    const _Float16* er = attnenc16 + ((size_t)b * 1024 + tid) * 256;
    float acc = 0.f;
    for (int a = 0; a < 256; a += 8) {
      v8h e8 = *(const v8h*)(er + a);
#pragma unroll
      for (int u = 0; u < 8; ++u)
        acc += v_s[a + u] * fast_tanh((float)e8[u] + d_s[a + u]);
    }
    float e = acc + v_b[0];
    if (enc_mask[(size_t)b * 1024 + tid] == 0.f) e = -1e9f;
    e_s[tid] = e;
    red[tid] = e;
  }
  __syncthreads();
  // softmax max-reduce over 1024
  for (int s = 512; s > 0; s >>= 1) {
    if (tid < s) red[tid] = fmaxf(red[tid], red[tid + s]);
    __syncthreads();
  }
  const float mx = red[0];
  __syncthreads();
  const float p = __expf(e_s[tid] - mx);
  e_s[tid] = p;
  red[tid] = p;
  __syncthreads();
  for (int s = 512; s > 0; s >>= 1) {
    if (tid < s) red[tid] += red[tid + s];
    __syncthreads();
  }
  const float inv = 1.0f / red[0];
  __syncthreads();
  // context[d] = inv * sum_t p_t * enc16[b,t,d]; quarter-split over t,
  // coalesced 512B rows across each 256-thread group.
  {
    const int d = tid & 255, q = tid >> 8;
    float acc = 0.f;
    const _Float16* eb = enc16 + ((size_t)b * 1024 + q * 256) * 256 + d;
#pragma unroll 8
    for (int tt = 0; tt < 256; ++tt) acc += e_s[q * 256 + tt] * (float)eb[(size_t)tt * 256];
    red[tid] = acc;
  }
  __syncthreads();
  if (tid < 256)
    ctx_t[b * 256 + tid] =
        (_Float16)((red[tid] + red[tid + 256] + red[tid + 512] + red[tid + 768]) * inv);
}

// ---------------------- per-step LSTM cell (WMMA gates) --------------------
// WG n computes gate N-tiles {n, n+16, n+32, n+48} -> i/f/g/o for hidden dims
// [16n,16n+16) are all local, so the cell update happens in the same kernel.
__global__ __launch_bounds__(256) void k_lstm(const _Float16* __restrict__ pre_t,
                                              const _Float16* __restrict__ ctx_t,
                                              const _Float16* __restrict__ h_in,
                                              const _Float16* __restrict__ Wcombt,
                                              const float* __restrict__ bias_comb,
                                              float* __restrict__ c_state,
                                              _Float16* __restrict__ h_out,
                                              _Float16* __restrict__ hall_t) {
  __shared__ float G[64][68];                    // 64 rows x (4 gates x 16 cols)
  const int nwg = blockIdx.x, tid = threadIdx.x;
  const int wave = tid >> 5, lane = tid & 31;
  const int mi = wave & 3, p = wave >> 2;        // wave -> (m-tile, gate-pair)
  const int j0 = nwg + 16 * (2 * p), j1 = nwg + 16 * (2 * p + 1);
  v8f acc0 = {}, acc1 = {};
  for (int kk = 0; kk < 24; ++kk) {
    const int k0 = kk * 32;
    const _Float16* Asrc;
    int koff;
    if (k0 < 256)      { Asrc = pre_t; koff = k0; }
    else if (k0 < 512) { Asrc = ctx_t; koff = k0 - 256; }
    else               { Asrc = h_in;  koff = k0 - 512; }
    v16h a  = load_a(Asrc + (size_t)(mi * 16) * 256 + koff, 256, lane);
    v16h b0 = load_b(Wcombt + (size_t)k0 * 1024 + j0 * 16, 1024, lane);
    v16h b1 = load_b(Wcombt + (size_t)k0 * 1024 + j1 * 16, 1024, lane);
    acc0 = wmma16(a, b0, acc0);
    acc1 = wmma16(a, b1, acc1);
  }
  {
    const int nlo = lane & 15, hi = lane >> 4;
#pragma unroll
    for (int r = 0; r < 8; ++r) {
      G[mi * 16 + r + 8 * hi][(2 * p) * 16 + nlo]     = acc0[r];
      G[mi * 16 + r + 8 * hi][(2 * p + 1) * 16 + nlo] = acc1[r];
    }
  }
  __syncthreads();
  for (int idx = tid; idx < 1024; idx += 256) {
    const int m = idx >> 4, cc = idx & 15;
    const int dim = nwg * 16 + cc;
    const float gi = G[m][cc]      + bias_comb[dim];
    const float gf = G[m][16 + cc] + bias_comb[256 + dim];
    const float gg = G[m][32 + cc] + bias_comb[512 + dim];
    const float go = G[m][48 + cc] + bias_comb[768 + dim];
    const float cold = c_state[m * 256 + dim];
    const float cn = sigm(gf) * cold + sigm(gi) * fast_tanh(gg);
    const float hn = sigm(go) * fast_tanh(cn);
    c_state[m * 256 + dim] = cn;
    const _Float16 hh = (_Float16)hn;
    h_out[m * 256 + dim] = hh;
    hall_t[m * 256 + dim] = hh;
  }
}

// ------------------- batched output heads (WMMA, at end) -------------------
__global__ __launch_bounds__(256) void k_heads(const _Float16* __restrict__ h_all,
                                               const _Float16* __restrict__ ctx_all,
                                               const _Float16* __restrict__ Wheadt,
                                               const float* __restrict__ biasH,
                                               float* __restrict__ mels,
                                               float* __restrict__ stops) {
  const int t = blockIdx.x;                       // 1000 blocks; rows R = t*64+m
  const size_t R0 = (size_t)t * 64;
  const int tid = threadIdx.x, wave = tid >> 5, lane = tid & 31;
  const int mi = wave >> 1, ntb = (wave & 1) * 3; // 24 tiles / 8 waves = 3 each
  v8f acc[3] = {};
  for (int kk = 0; kk < 16; ++kk) {
    const int k0 = kk * 32;
    const _Float16* Asrc = (k0 < 256) ? (h_all + R0 * 256 + k0)
                                      : (ctx_all + R0 * 256 + (k0 - 256));
    v16h a = load_a(Asrc + (size_t)(mi * 16) * 256, 256, lane);
#pragma unroll
    for (int q = 0; q < 3; ++q) {
      v16h b = load_b(Wheadt + (size_t)k0 * 96 + (ntb + q) * 16, 96, lane);
      acc[q] = wmma16(a, b, acc[q]);
    }
  }
  const int nlo = lane & 15, hi = lane >> 4;
  for (int q = 0; q < 3; ++q) {
    const int n = (ntb + q) * 16 + nlo;
    const float bb = biasH[n];
#pragma unroll
    for (int r = 0; r < 8; ++r) {
      const int m = mi * 16 + r + 8 * hi;
      const float v = acc[q][r] + bb;
      if (n < 80)       mels[((size_t)m * T_OUTS + t) * 80 + n] = v;
      else if (n == 80) stops[(size_t)m * T_OUTS + t] = v;
    }
  }
}

// ---------------------------------------------------------------------------

extern "C" void kernel_launch(void* const* d_in, const int* in_sizes, int n_in,
                              void* d_out, int out_size, void* d_ws, size_t ws_size,
                              hipStream_t stream) {
  (void)in_sizes; (void)n_in; (void)out_size; (void)ws_size;
  const float* enc      = (const float*)d_in[0];
  const float* tmel     = (const float*)d_in[1];
  const float* W_enc    = (const float*)d_in[2];
  const float* b_enc    = (const float*)d_in[3];
  const float* W_dec    = (const float*)d_in[4];
  const float* b_dec    = (const float*)d_in[5];
  const float* v_w      = (const float*)d_in[6];
  const float* v_b      = (const float*)d_in[7];
  const float* pre_W1   = (const float*)d_in[8];
  const float* pre_b1   = (const float*)d_in[9];
  const float* pre_W2   = (const float*)d_in[10];
  const float* pre_b2   = (const float*)d_in[11];
  const float* W_ih     = (const float*)d_in[12];
  const float* W_hh     = (const float*)d_in[13];
  const float* b_ih     = (const float*)d_in[14];
  const float* b_hh     = (const float*)d_in[15];
  const float* mel_W    = (const float*)d_in[16];
  const float* mel_b    = (const float*)d_in[17];
  const float* stop_W   = (const float*)d_in[18];
  const float* stop_b   = (const float*)d_in[19];
  const float* enc_mask = (const float*)d_in[20];

  char* ws = (char*)d_ws;
  size_t off = 0;
  auto alloc = [&](size_t bytes) -> void* {
    off = (off + 255) & ~(size_t)255;
    void* p = ws + off;
    off += bytes;
    return p;
  };
  _Float16* enc16     = (_Float16*)alloc((size_t)64 * 1024 * 256 * 2);
  _Float16* attnenc16 = (_Float16*)alloc((size_t)64 * 1024 * 256 * 2);
  _Float16* pre_all   = (_Float16*)alloc((size_t)T_OUTS * 64 * 256 * 2);
  _Float16* h_all     = (_Float16*)alloc((size_t)T_OUTS * 64 * 256 * 2);
  _Float16* ctx_all   = (_Float16*)alloc((size_t)T_OUTS * 64 * 256 * 2);
  _Float16* Wcombt    = (_Float16*)alloc((size_t)768 * 1024 * 2);
  float*    bias_comb = (float*)alloc(1024 * 4);
  _Float16* Wheadt    = (_Float16*)alloc((size_t)512 * 96 * 2);
  float*    biasH     = (float*)alloc(96 * 4);
  _Float16* W1t       = (_Float16*)alloc((size_t)96 * 256 * 2);
  _Float16* W2t       = (_Float16*)alloc((size_t)256 * 256 * 2);
  _Float16* Wenct     = (_Float16*)alloc((size_t)256 * 256 * 2);
  _Float16* Wdec16    = (_Float16*)alloc((size_t)256 * 256 * 2);
  _Float16* h16a      = (_Float16*)alloc((size_t)64 * 256 * 2);
  _Float16* h16b      = (_Float16*)alloc((size_t)64 * 256 * 2);
  float*    c_state   = (float*)alloc((size_t)64 * 256 * 4);

  float* mels  = (float*)d_out;
  float* stops = mels + (size_t)64 * T_OUTS * 80;

  // --- one-time weight prep / conversions ---
  k_cvt   <<<256, 256, 0, stream>>>(W_dec, Wdec16, 256 * 256);
  k_t256  <<<256, 256, 0, stream>>>(W_enc, Wenct);
  k_t256  <<<256, 256, 0, stream>>>(pre_W2, W2t);
  k_w1t   <<<96, 256, 0, stream>>>(pre_W1, W1t);
  k_wcomb <<<3072, 256, 0, stream>>>(W_ih, W_hh, b_ih, b_hh, Wcombt, bias_comb);
  k_whead <<<192, 256, 0, stream>>>(mel_W, stop_W, mel_b, stop_b, Wheadt, biasH);
  k_init  <<<64, 256, 0, stream>>>(h16a, c_state);

  // --- batched precompute: encoder projection + prenet (WMMA GEMMs) ---
  k_encproj<<<1024, 256, 0, stream>>>(enc, b_enc, Wenct, enc16, attnenc16);
  k_prenet <<<T_OUTS, 256, 0, stream>>>(tmel, pre_b1, pre_b2, W1t, W2t, pre_all);

  // --- sequential decode: 2 kernels per step, h double-buffered ---
  _Float16* hbuf[2] = {h16a, h16b};
  for (int t = 0; t < T_OUTS; ++t) {
    _Float16* h_in   = hbuf[t & 1];
    _Float16* h_out  = hbuf[(t + 1) & 1];
    _Float16* ctx_t  = ctx_all + (size_t)t * 64 * 256;
    _Float16* pre_t  = pre_all + (size_t)t * 64 * 256;
    _Float16* hall_t = h_all + (size_t)t * 64 * 256;
    k_attn<<<64, 1024, 0, stream>>>(attnenc16, enc16, Wdec16, h_in, b_dec, v_w,
                                    v_b, enc_mask, ctx_t);
    k_lstm<<<16, 256, 0, stream>>>(pre_t, ctx_t, h_in, Wcombt, bias_comb,
                                   c_state, h_out, hall_t);
  }

  // --- batched output heads (WMMA GEMM over all 64000 rows) ---
  k_heads<<<T_OUTS, 256, 0, stream>>>(h_all, ctx_all, Wheadt, biasH, mels, stops);
}
